// MultiQueryAttentionLayer_39702677684913
// MI455X (gfx1250) — matrix operand order repaired and checked by reading, not compiled, not run on hardware
//
#include <hip/hip_runtime.h>
#include <hip/hip_bf16.h>

typedef __attribute__((ext_vector_type(16))) _Float16 v16h;
typedef __attribute__((ext_vector_type(8)))  float    v8f;

#define HID   1024
#define NHEAD 16
#define DHEAD 64
#define BATCH 2
#define SEQ   2048
#define MTOT  (BATCH * SEQ)   // 4096 token rows

// A-fragment K index pattern (16-bit A 16x32, ISA 7.12.2)
__device__ __forceinline__ int kkA(int vv, int hi) {
    return (vv >> 2) * 16 + hi * 8 + (vv & 3) * 2;
}
// B-fragment K index pattern (16-bit B 32x16, same K order as A)
__device__ __forceinline__ int kkB(int vv, int hi) {
    return (vv >> 2) * 16 + hi * 8 + (vv & 3) * 2;
}

// ---------------------------------------------------------------------------
// Convert + transpose f32 weight [K,N] -> f16 WT [N,K]
// ---------------------------------------------------------------------------
__global__ void wtr_kernel(const float* __restrict__ W, _Float16* __restrict__ WT,
                           int K, int N) {
    int i = blockIdx.x * 256 + threadIdx.x;
    if (i >= K * N) return;
    int k = i / N, n = i % N;              // coalesced read over n
    WT[(size_t)n * K + k] = (_Float16)W[i];
}

// ---------------------------------------------------------------------------
// Transpose v_h [B*S, 64] -> vT [(b*64+d)*S + s]
// ---------------------------------------------------------------------------
__global__ void vtr_kernel(const _Float16* __restrict__ v_h, _Float16* __restrict__ vT) {
    int i = blockIdx.x * 256 + threadIdx.x;
    if (i >= MTOT * DHEAD) return;
    int d = i & 63;
    int s = (i >> 6) & (SEQ - 1);
    int b = i >> 17;
    vT[((size_t)(b * DHEAD + d)) * SEQ + s] = v_h[i];
}

// ---------------------------------------------------------------------------
// GEMM: C[M,N] = A[M,K](f32) * WT[N,K]^T(f16) + bias
// 8 waves/block; each wave computes a 16x64 strip (4 accumulators) so the
// A fragment is reused 4x per K-step (12 b128 loads per 4 WMMAs).
// F32OUT=true -> f32 output (+bias); false -> f16 output (+bias).
// ---------------------------------------------------------------------------
template <bool F32OUT>
__global__ __launch_bounds__(256) void gemm_kernel(
    const float* __restrict__ A, const _Float16* __restrict__ WT,
    const float* __restrict__ bias, void* __restrict__ outv,
    int M, int N, int K) {

    const int wave = threadIdx.x >> 5;
    const int lane = threadIdx.x & 31;
    const int col  = lane & 15;
    const int hi   = lane >> 4;

    const int ngroups = N >> 6;                 // 64-wide N strips
    const int strip   = blockIdx.x * 8 + wave;
    if (strip >= (M >> 4) * ngroups) return;
    const int m0 = (strip / ngroups) * 16;
    const int n0 = (strip % ngroups) * 64;

    const float* arow = A + (size_t)(m0 + col) * K;
    const _Float16* brow[4];
#pragma unroll
    for (int j = 0; j < 4; ++j)
        brow[j] = WT + (size_t)(n0 + j * 16 + col) * K;

    v8f acc[4] = {{}, {}, {}, {}};
    for (int k0 = 0; k0 < K; k0 += 32) {
        v16h af;
#pragma unroll
        for (int vv = 0; vv < 8; ++vv) {
            int ka = k0 + kkA(vv, hi);
            af[2 * vv]     = (_Float16)arow[ka];
            af[2 * vv + 1] = (_Float16)arow[ka + 1];
        }
#pragma unroll
        for (int j = 0; j < 4; ++j) {
            v16h bf;
#pragma unroll
            for (int vv = 0; vv < 8; ++vv) {
                int kb = k0 + kkB(vv, hi);
                bf[2 * vv]     = brow[j][kb];
                bf[2 * vv + 1] = brow[j][kb + 1];
            }
            acc[j] = __builtin_amdgcn_wmma_f32_16x16x32_f16(
                false, af, false, bf, (short)0, acc[j], false, false);
        }
    }

#pragma unroll
    for (int j = 0; j < 4; ++j) {
        const float bv = bias[n0 + j * 16 + col];
#pragma unroll
        for (int r = 0; r < 8; ++r) {
            int m = m0 + r + hi * 8;
            float val = acc[j][r] + bv;
            size_t idx = (size_t)m * N + n0 + j * 16 + col;
            if (F32OUT) ((float*)outv)[idx] = val;
            else        ((_Float16*)outv)[idx] = (_Float16)val;
        }
    }
}

// ---------------------------------------------------------------------------
// Fused attention: per block = (b, h, 16-query tile).
//   Phase 1: energy strip 16 x 2048 into LDS (WMMA QK^T, 8 waves x 16 tiles)
//   Phase 2: softmax in LDS, write normalized attn to global (only HBM write)
//   Phase 3: ctx = attn * V via WMMA from LDS, reduce across waves, write f32 ctx
// LDS: 128KB strip + ~5KB scratch < 320KB/WGP.
// ---------------------------------------------------------------------------
__global__ __launch_bounds__(256) void attn_kernel(
    const _Float16* __restrict__ q_h,   // [B*S, HID] f16
    const _Float16* __restrict__ k_h,   // [B*S, 64]  f16
    const _Float16* __restrict__ vT,    // [B*64, S]  f16
    float* __restrict__ attn,           // [B,H,S,S]  f32
    float* __restrict__ ctx) {          // [B*S, HID] f32

    __shared__ float sm_e[16][SEQ];     // 128 KB energy / attn strip
    __shared__ float sm_red[16][16];
    __shared__ float sm_max[16];
    __shared__ float sm_rinv[16];
    __shared__ float sm_ctx[16][DHEAD]; // 4 KB

    const int wave = threadIdx.x >> 5;
    const int lane = threadIdx.x & 31;
    const int col  = lane & 15;
    const int hi   = lane >> 4;

    const int qt = blockIdx.x;          // 0..127
    const int h  = blockIdx.y;          // 0..15
    const int bb = blockIdx.z;          // 0..1
    const int q0 = qt * 16;

    // ---- Phase 1: energy = (q . k^T) / 8 -----------------------------------
    const _Float16* qrow = q_h + ((size_t)(bb * SEQ + q0 + col)) * HID + h * DHEAD;
    v16h qa0, qa1;
#pragma unroll
    for (int vv = 0; vv < 8; ++vv) {
        int ka = kkA(vv, hi);
        qa0[2 * vv]     = qrow[ka];
        qa0[2 * vv + 1] = qrow[ka + 1];
        qa1[2 * vv]     = qrow[32 + ka];
        qa1[2 * vv + 1] = qrow[32 + ka + 1];
    }

    const float rscale = 0.125f; // 1/sqrt(64)
    for (int kt = wave; kt < SEQ / 16; kt += 8) {
        const int kbase = kt * 16;
        const _Float16* krow = k_h + ((size_t)(bb * SEQ + kbase + col)) * DHEAD;
        v16h kb0, kb1;
#pragma unroll
        for (int vv = 0; vv < 8; ++vv) {
            int kb = kkB(vv, hi);
            kb0[2 * vv]     = krow[kb];
            kb0[2 * vv + 1] = krow[kb + 1];
            kb1[2 * vv]     = krow[32 + kb];
            kb1[2 * vv + 1] = krow[32 + kb + 1];
        }
        v8f acc = {};
        acc = __builtin_amdgcn_wmma_f32_16x16x32_f16(false, qa0, false, kb0, (short)0, acc, false, false);
        acc = __builtin_amdgcn_wmma_f32_16x16x32_f16(false, qa1, false, kb1, (short)0, acc, false, false);
#pragma unroll
        for (int r = 0; r < 8; ++r)
            sm_e[r + hi * 8][kbase + col] = acc[r] * rscale;
    }
    __syncthreads();

    // ---- Phase 2: softmax over rows of the 16x2048 strip -------------------
    const int tr = threadIdx.x >> 4;    // row 0..15
    const int tc = threadIdx.x & 15;    // 16 threads per row
    float mx = -3.0e38f;
    for (int c = tc; c < SEQ; c += 16) mx = fmaxf(mx, sm_e[tr][c]);
    sm_red[tr][tc] = mx;
    __syncthreads();
    if (threadIdx.x < 16) {
        float mm = sm_red[threadIdx.x][0];
#pragma unroll
        for (int j = 1; j < 16; ++j) mm = fmaxf(mm, sm_red[threadIdx.x][j]);
        sm_max[threadIdx.x] = mm;
    }
    __syncthreads();
    const float rm = sm_max[tr];
    float ssum = 0.f;
    for (int c = tc; c < SEQ; c += 16) {
        float ev = __expf(sm_e[tr][c] - rm);
        sm_e[tr][c] = ev;
        ssum += ev;
    }
    sm_red[tr][tc] = ssum;
    __syncthreads();
    if (threadIdx.x < 16) {
        float tot = 0.f;
#pragma unroll
        for (int j = 0; j < 16; ++j) tot += sm_red[threadIdx.x][j];
        sm_rinv[threadIdx.x] = 1.0f / tot;
    }
    __syncthreads();

    // normalize in LDS + single coalesced write of attn; zero ctx accumulator
    for (int i = threadIdx.x; i < 16 * DHEAD; i += 256)
        sm_ctx[i >> 6][i & 63] = 0.f;
    float* attn_base = attn + (((size_t)(bb * NHEAD + h) * SEQ + q0) * SEQ);
    for (int i = threadIdx.x; i < 16 * SEQ; i += 256) {
        int r = i >> 11, c = i & (SEQ - 1);
        float vn = sm_e[r][c] * sm_rinv[r];
        sm_e[r][c] = vn;
        attn_base[((size_t)r << 11) + c] = vn;
    }
    __syncthreads();

    // ---- Phase 3: ctx = attn * V -------------------------------------------
    v8f acc[4] = {{}, {}, {}, {}};
    for (int kp = wave; kp < SEQ / 32; kp += 8) {
        const int kbase = kp * 32;
        v16h af;
#pragma unroll
        for (int vv = 0; vv < 8; ++vv) {
            int ka = kbase + kkA(vv, hi);
            af[2 * vv]     = (_Float16)sm_e[col][ka];
            af[2 * vv + 1] = (_Float16)sm_e[col][ka + 1];
        }
#pragma unroll
        for (int j = 0; j < 4; ++j) {
            const _Float16* vrow = vT + ((size_t)(bb * DHEAD + j * 16 + col)) * SEQ + kbase;
            v16h bf;
#pragma unroll
            for (int vv = 0; vv < 8; ++vv) {
                int kb = kkB(vv, hi);
                bf[2 * vv]     = vrow[kb];
                bf[2 * vv + 1] = vrow[kb + 1];
            }
            acc[j] = __builtin_amdgcn_wmma_f32_16x16x32_f16(
                false, af, false, bf, (short)0, acc[j], false, false);
        }
    }
#pragma unroll
    for (int j = 0; j < 4; ++j)
#pragma unroll
        for (int r = 0; r < 8; ++r)
            atomicAdd(&sm_ctx[r + hi * 8][j * 16 + col], acc[j][r]);
    __syncthreads();

    float* ctxp = ctx + ((size_t)(bb * SEQ + q0)) * HID + h * DHEAD;
    for (int i = threadIdx.x; i < 16 * DHEAD; i += 256) {
        int r = i >> 6, dd = i & 63;
        ctxp[(size_t)r * HID + dd] = sm_ctx[r][dd];
    }
}

// ---------------------------------------------------------------------------
extern "C" void kernel_launch(void* const* d_in, const int* in_sizes, int n_in,
                              void* d_out, int out_size, void* d_ws, size_t ws_size,
                              hipStream_t stream) {
    const float* query   = (const float*)d_in[0];
    const float* key     = (const float*)d_in[1];
    const float* value   = (const float*)d_in[2];
    const float* wq_w    = (const float*)d_in[3];
    const float* wq_b    = (const float*)d_in[4];
    const float* wk_w    = (const float*)d_in[5];
    const float* wk_b    = (const float*)d_in[6];
    const float* wv_w    = (const float*)d_in[7];
    const float* wv_b    = (const float*)d_in[8];
    const float* dense_w = (const float*)d_in[9];
    const float* dense_b = (const float*)d_in[10];

    float* out  = (float*)d_out;                       // [B,S,HID]
    float* attn = out + (size_t)MTOT * HID;            // [B,H,S,S]

    // workspace carve-up (256B aligned)
    char* w = (char*)d_ws;
    size_t off = 0;
    auto alloc = [&](size_t bytes) -> void* {
        void* p = w + off;
        off += (bytes + 255) & ~(size_t)255;
        return p;
    };
    _Float16* wqT = (_Float16*)alloc((size_t)HID * HID * 2);
    _Float16* wkT = (_Float16*)alloc((size_t)HID * DHEAD * 2);
    _Float16* wvT = (_Float16*)alloc((size_t)HID * DHEAD * 2);
    _Float16* dwT = (_Float16*)alloc((size_t)HID * HID * 2);
    _Float16* q_h = (_Float16*)alloc((size_t)MTOT * HID * 2);
    _Float16* k_h = (_Float16*)alloc((size_t)MTOT * DHEAD * 2);
    _Float16* v_h = (_Float16*)alloc((size_t)MTOT * DHEAD * 2);
    _Float16* vT  = (_Float16*)alloc((size_t)MTOT * DHEAD * 2);
    float*    ctx = (float*)alloc((size_t)MTOT * HID * 4);

    // 1) weight transposes (f32 -> f16 [N,K])
    wtr_kernel<<<(HID * HID + 255) / 256, 256, 0, stream>>>(wq_w, wqT, HID, HID);
    wtr_kernel<<<(HID * DHEAD + 255) / 256, 256, 0, stream>>>(wk_w, wkT, HID, DHEAD);
    wtr_kernel<<<(HID * DHEAD + 255) / 256, 256, 0, stream>>>(wv_w, wvT, HID, DHEAD);
    wtr_kernel<<<(HID * HID + 255) / 256, 256, 0, stream>>>(dense_w, dwT, HID, HID);

    // 2) projections (WMMA GEMM, f16 outputs); each wave owns a 16x64 strip
    gemm_kernel<false><<<(MTOT / 16) * (HID / 64) / 8, 256, 0, stream>>>(
        query, wqT, wq_b, q_h, MTOT, HID, HID);
    gemm_kernel<false><<<(MTOT / 16) * (DHEAD / 64) / 8, 256, 0, stream>>>(
        key, wkT, wk_b, k_h, MTOT, DHEAD, HID);
    gemm_kernel<false><<<(MTOT / 16) * (DHEAD / 64) / 8, 256, 0, stream>>>(
        value, wvT, wv_b, v_h, MTOT, DHEAD, HID);

    // 3) v transpose for B-fragment-friendly layout
    vtr_kernel<<<(MTOT * DHEAD + 255) / 256, 256, 0, stream>>>(v_h, vT);

    // 4) fused attention (energy + softmax + attn write + ctx)
    dim3 agrid(SEQ / 16, NHEAD, BATCH);
    attn_kernel<<<agrid, 256, 0, stream>>>(q_h, k_h, vT, attn, ctx);

    // 5) dense output layer (WMMA GEMM, f32 out + bias)
    gemm_kernel<true><<<(MTOT / 16) * (HID / 64) / 8, 256, 0, stream>>>(
        ctx, dwT, dense_b, out, MTOT, HID, HID);
}
